// MultiHeadAttention_88330297409596
// MI455X (gfx1250) — matrix operand order repaired and checked by reading, not compiled, not run on hardware
//
#include <hip/hip_runtime.h>
#include <hip/hip_bf16.h>

#define DIM      512
#define HEADS    8
#define HEAD_DIM 64
#define BATCH    8
#define SEQ      2048
#define NQKV     1536                 // 3*DIM
#define M_ROWS   (BATCH * SEQ)        // 16384

typedef __attribute__((ext_vector_type(16))) __bf16 v16bf;
typedef __attribute__((ext_vector_type(8)))  float  v8f;
typedef unsigned short u16;

union ABReg { v16bf v; uint4 q[2]; };

// Two 16B chunks -> one 8-VGPR WMMA operand (A and B: contiguous 32B per lane, k = 16*hi + s)
__device__ __forceinline__ v16bf load_ab(const u16* p0, const u16* p1) {
  ABReg r;
  r.q[0] = *(const uint4*)p0;
  r.q[1] = *(const uint4*)p1;
  return r.v;
}

__device__ __forceinline__ u16 f2bf(float f) {           // RNE f32 -> bf16
  unsigned u = __float_as_uint(f);
  u += 0x7fffu + ((u >> 16) & 1u);
  return (u16)(u >> 16);
}

__device__ __forceinline__ v8f wmma_bf16(v16bf a, v16bf b, v8f c) {
  return __builtin_amdgcn_wmma_f32_16x16x32_bf16(false, a, false, b, (short)0, c,
                                                 false, false);
}

__device__ __forceinline__ float elu1(float x) { return x > 0.0f ? x : expm1f(x); }

// ---------------------------------------------------------------- convert
__global__ __launch_bounds__(256) void cvt_f32_bf16(const float* __restrict__ in,
                                                    u16* __restrict__ out, int n) {
  int i = blockIdx.x * blockDim.x + threadIdx.x;
  int stride = gridDim.x * blockDim.x;
  for (; i < n; i += stride) out[i] = f2bf(in[i]);
}

// ---------------------------------------------------------------- QKV GEMM
// block = 128(M) x 64(N); 8 waves share a W slab staged per k-step in LDS.
// out[m,n] = sum_k x[m,k]*Wqkv[n,k] + bqkv[n]; n -> (h=n/192, d=(n%192)/3, which=n%3)
// q -> [b,h,s,d] bf16 ; k -> [b,h,s,d] bf16 ; v -> elu, transposed [b,h,d,s] bf16
__global__ __launch_bounds__(256) void qkv_gemm(const u16* __restrict__ xbf,
                                                const u16* __restrict__ wbf,
                                                const float* __restrict__ bqkv,
                                                u16* __restrict__ qbf,
                                                u16* __restrict__ kbf,
                                                u16* __restrict__ vtbf) {
  __shared__ __align__(16) u16 Wlds[2][64][32];   // double-buffered 64n x 32k slab
  const int tid  = threadIdx.x;
  const int w    = tid >> 5;
  const int lane = tid & 31;
  const int row  = lane & 15;
  const int hi   = lane >> 4;
  const int n0   = (blockIdx.x % 24) * 64;        // shared by all waves in block
  const int m0   = (blockIdx.x / 24) * 128 + w * 16;

  const u16* arow = xbf + (size_t)(m0 + row) * DIM;
  auto loadA = [&](int k0) {
    return load_ab(arow + k0 + hi * 16, arow + k0 + hi * 16 + 8);
  };
  // cooperative W staging: one b128 per thread per step
  const int sn = tid >> 2;                        // 0..63 slab row (n)
  const int sc = (tid & 3) * 8;                   // 8-element chunk within 32 k
  const u16* wsrc = wbf + (size_t)(n0 + sn) * DIM + sc;
  *(uint4*)&Wlds[0][sn][sc] = *(const uint4*)wsrc;   // stage k-step 0

  const v8f z = {0.f,0.f,0.f,0.f,0.f,0.f,0.f,0.f};
  v8f acc[4] = {z, z, z, z};
  v16bf a[2];
  a[0] = loadA(0);

  #pragma unroll
  for (int s = 0; s < 16; ++s) {
    const int cur = s & 1, nxt = cur ^ 1;
    __syncthreads();                              // Wlds[cur] visible to all waves
    uint4 wreg;
    if (s < 15) {                                 // issue next-step loads early
      wreg   = *(const uint4*)(wsrc + 32 * (s + 1));
      a[nxt] = loadA(32 * (s + 1));
    }
    #pragma unroll
    for (int j = 0; j < 4; ++j) {
      const u16* bp = &Wlds[cur][j * 16 + row][hi * 16];
      acc[j] = wmma_bf16(a[cur], load_ab(bp, bp + 8), acc[j]);
    }
    if (s < 15) *(uint4*)&Wlds[nxt][sn][sc] = wreg;   // store after compute
  }

  #pragma unroll
  for (int j = 0; j < 4; ++j) {
    const int n     = n0 + j * 16 + row;          // this lane's output column
    const int h     = n / 192;
    const int rem   = n % 192;
    const int d     = rem / 3;
    const int which = rem % 3;
    const float bias = bqkv[n];
    #pragma unroll
    for (int r = 0; r < 8; ++r) {
      const int mg  = m0 + hi * 8 + r;            // C layout: M = r + hi*8
      const int bb2 = mg >> 11;                   // /SEQ
      const int ss  = mg & (SEQ - 1);
      float val = acc[j][r] + bias;
      if (which == 0) {
        qbf[(((size_t)bb2 * HEADS + h) * SEQ + ss) * HEAD_DIM + d] = f2bf(val);
      } else if (which == 1) {
        kbf[(((size_t)bb2 * HEADS + h) * SEQ + ss) * HEAD_DIM + d] = f2bf(val);
      } else {
        vtbf[(((size_t)bb2 * HEADS + h) * HEAD_DIM + d) * SEQ + ss] = f2bf(elu1(val));
      }
    }
  }
}

// ---------------------------------------------------------------- flash attention
// block owns one (b,h) + 8 consecutive 16-query tiles; K/V tiles staged once per
// block into double-buffered LDS (8x less L2 traffic than per-wave streaming).
// Per iter: sync -> issue next-tile global loads -> compute from LDS[cur]
// (scores, online softmax, P-bounce, PV) -> ds_store staged regs -> repeat.
__global__ __launch_bounds__(256) void attention(const u16* __restrict__ qbf,
                                                 const u16* __restrict__ kbf,
                                                 const u16* __restrict__ vtbf,
                                                 const unsigned char* __restrict__ mask,
                                                 u16* __restrict__ aobf) {
  __shared__ __align__(16) u16 Kt[2][32][64];     // 32 keys x 64 d
  __shared__ __align__(16) u16 Vt[2][64][32];     // 64 d x 32 keys (transposed V)
  __shared__ __align__(16) u16 Plds[8][16][32];   // per-wave prob tile (C->A re-stripe)
  const int tid  = threadIdx.x;
  const int w    = tid >> 5;
  const int lane = tid & 31;
  const int row  = lane & 15;
  const int hi   = lane >> 4;
  const int bhIdx = blockIdx.x >> 4;              // 0..63, uniform per block
  const int qblk  = blockIdx.x & 15;
  const int b     = bhIdx >> 3;
  const int h     = bhIdx & 7;
  const int m0    = (qblk * 8 + w) * 16;

  const size_t bh = (size_t)b * HEADS + h;
  const u16* qp = qbf + bh * SEQ * HEAD_DIM;
  const u16* kp = kbf + bh * SEQ * HEAD_DIM;
  const u16* vp = vtbf + bh * HEAD_DIM * SEQ;
  const unsigned char* mp = mask + (size_t)b * SEQ;

  const u16* qrow = qp + (size_t)(m0 + row) * HEAD_DIM;
  const v16bf aq0 = load_ab(qrow + hi * 16,      qrow + hi * 16 + 8);       // d 0..31
  const v16bf aq1 = load_ab(qrow + 32 + hi * 16, qrow + 32 + hi * 16 + 8);  // d 32..63

  // cooperative staging: one b128 of K and one of V per thread per tile
  const int krow = tid >> 3, kc = (tid & 7) * 8;  // K: 32 rows x 64 elems
  const int vrow = tid >> 2, vc = (tid & 3) * 8;  // V: 64 rows x 32 elems
  const u16* ksrc = kp + (size_t)krow * HEAD_DIM + kc;
  const u16* vsrc = vp + (size_t)vrow * SEQ + vc;
  *(uint4*)&Kt[0][krow][kc] = *(const uint4*)ksrc;           // stage tile 0
  *(uint4*)&Vt[0][vrow][vc] = *(const uint4*)vsrc;

  const v8f z = {0.f,0.f,0.f,0.f,0.f,0.f,0.f,0.f};
  v8f o[4] = {z, z, z, z};
  float mrun[8], lrun[8];
  #pragma unroll
  for (int r = 0; r < 8; ++r) { mrun[r] = -1e30f; lrun[r] = 0.f; }

  #pragma unroll 1
  for (int it = 0; it < SEQ / 32; ++it) {
    const int kt  = it * 32;
    const int cur = it & 1, nxt = cur ^ 1;
    __syncthreads();                              // staged tile [cur] ready

    const bool more = (it + 1 < SEQ / 32);
    uint4 kreg, vreg;
    if (more) {                                   // issue next-tile loads early
      kreg = *(const uint4*)(ksrc + (size_t)(kt + 32) * HEAD_DIM);
      vreg = *(const uint4*)(vsrc + kt + 32);
      if (it + 2 < SEQ / 32)                      // warm L0 for the tile after
        __builtin_prefetch(ksrc + (size_t)(kt + 64) * HEAD_DIM, 0, 0);
    }

    // ---- scores: two 16x16 tiles from LDS K (B operand: 32B/lane, ds_load_b128)
    const u16 (*Kc)[64] = Kt[cur];
    const u16 (*Vc)[32] = Vt[cur];
    v8f s0 = z, s1 = z;
    { const u16* p = &Kc[row][hi * 16];           s0 = wmma_bf16(aq0, load_ab(p, p + 8), s0); }
    { const u16* p = &Kc[row][32 + hi * 16];      s0 = wmma_bf16(aq1, load_ab(p, p + 8), s0); }
    { const u16* p = &Kc[16 + row][hi * 16];      s1 = wmma_bf16(aq0, load_ab(p, p + 8), s1); }
    { const u16* p = &Kc[16 + row][32 + hi * 16]; s1 = wmma_bf16(aq1, load_ab(p, p + 8), s1); }

    const float bias0 = mp[kt + row]      ? -1e30f : 0.0f;   // lane owns one key column
    const float bias1 = mp[kt + 16 + row] ? -1e30f : 0.0f;

    float p0[8], p1[8], sf[8];
    #pragma unroll
    for (int r = 0; r < 8; ++r) {
      float a0 = s0[r] * 0.125f + bias0;          // 1/sqrt(64)
      float a1 = s1[r] * 0.125f + bias1;
      float mx = fmaxf(a0, a1);                   // row max over 16-lane half
      mx = fmaxf(mx, __shfl_xor(mx, 1));
      mx = fmaxf(mx, __shfl_xor(mx, 2));
      mx = fmaxf(mx, __shfl_xor(mx, 4));
      mx = fmaxf(mx, __shfl_xor(mx, 8));
      float mn = fmaxf(mrun[r], mx);
      sf[r] = __expf(mrun[r] - mn);
      mrun[r] = mn;
      p0[r] = __expf(a0 - mn);
      p1[r] = __expf(a1 - mn);
      float rs = p0[r] + p1[r];
      rs += __shfl_xor(rs, 1);
      rs += __shfl_xor(rs, 2);
      rs += __shfl_xor(rs, 4);
      rs += __shfl_xor(rs, 8);
      lrun[r] = lrun[r] * sf[r] + rs;
    }
    #pragma unroll
    for (int j = 0; j < 4; ++j)
      #pragma unroll
      for (int r = 0; r < 8; ++r) o[j][r] *= sf[r];

    // ---- C-layout probs -> LDS -> A-layout operand
    #pragma unroll
    for (int r = 0; r < 8; ++r) {
      Plds[w][hi * 8 + r][row]      = f2bf(p0[r]);
      Plds[w][hi * 8 + r][16 + row] = f2bf(p1[r]);
    }
    asm volatile("s_wait_dscnt 0" ::: "memory");  // same-wave LDS RAW
    const u16* pr = &Plds[w][row][hi * 16];
    v16bf ap = load_ab(pr, pr + 8);

    // ---- PV: 4 d-tiles from LDS V
    #pragma unroll
    for (int j = 0; j < 4; ++j) {
      const u16* p = &Vc[j * 16 + row][hi * 16];
      o[j] = wmma_bf16(ap, load_ab(p, p + 8), o[j]);
    }

    // ---- store staged regs into the other buffer (after compute; loads have landed)
    if (more) {
      *(uint4*)&Kt[nxt][krow][kc] = kreg;
      *(uint4*)&Vt[nxt][vrow][vc] = vreg;
    }
  }

  #pragma unroll
  for (int j = 0; j < 4; ++j) {
    const int col = h * HEAD_DIM + j * 16 + row;  // out reshaped [b,s,h*64+d]
    #pragma unroll
    for (int r = 0; r < 8; ++r) {
      const int ss = m0 + hi * 8 + r;
      aobf[((size_t)b * SEQ + ss) * DIM + col] = f2bf(o[j][r] / lrun[r]);
    }
  }
}

// ---------------------------------------------------------------- out-proj + LN + ELU
// block owns 16 full rows -> LayerNorm fused via 32KB LDS
__global__ __launch_bounds__(256) void outproj_ln(const u16* __restrict__ aobf,
                                                  const u16* __restrict__ wobf,
                                                  const float* __restrict__ bout,
                                                  const float* __restrict__ gamma,
                                                  const float* __restrict__ beta,
                                                  float* __restrict__ out) {
  __shared__ __align__(16) float Sld[16][DIM];
  const int w    = threadIdx.x >> 5;
  const int lane = threadIdx.x & 31;
  const int row  = lane & 15;
  const int hi   = lane >> 4;
  const int m0   = blockIdx.x * 16;

  const u16* arow = aobf + (size_t)(m0 + row) * DIM;
  auto loadA = [&](int k0) {
    return load_ab(arow + k0 + hi * 16, arow + k0 + hi * 16 + 8);
  };
  auto loadB = [&](int j, int k0) {
    const u16* bp = wobf + (size_t)(w * 64 + j * 16 + row) * DIM + k0 + hi * 16;
    return load_ab(bp, bp + 8);
  };

  const v8f z = {0.f,0.f,0.f,0.f,0.f,0.f,0.f,0.f};
  v8f acc[4] = {z, z, z, z};

  v16bf ab[2];
  v16bf bb[2][4];
  ab[0] = loadA(0);
  #pragma unroll
  for (int j = 0; j < 4; ++j) bb[0][j] = loadB(j, 0);
  #pragma unroll
  for (int s = 0; s < 16; ++s) {
    const int cur = s & 1, nxt = cur ^ 1;
    if (s < 15) {
      ab[nxt] = loadA(32 * (s + 1));
      #pragma unroll
      for (int j = 0; j < 4; ++j) bb[nxt][j] = loadB(j, 32 * (s + 1));
    }
    #pragma unroll
    for (int j = 0; j < 4; ++j) acc[j] = wmma_bf16(ab[cur], bb[cur][j], acc[j]);
  }

  #pragma unroll
  for (int j = 0; j < 4; ++j) {
    const int n = w * 64 + j * 16 + row;
    const float bb2 = bout[n];
    #pragma unroll
    for (int r = 0; r < 8; ++r) Sld[hi * 8 + r][n] = acc[j][r] + bb2;
  }
  __syncthreads();

  const int r   = threadIdx.x >> 4;      // row 0..15
  const int seg = threadIdx.x & 15;      // 32-col segment
  const float* rp = &Sld[r][seg * 32];
  float s = 0.f, s2 = 0.f;
  #pragma unroll
  for (int c = 0; c < 32; ++c) { float v = rp[c]; s += v; s2 += v * v; }
  s  += __shfl_xor(s, 1);  s  += __shfl_xor(s, 2);
  s  += __shfl_xor(s, 4);  s  += __shfl_xor(s, 8);
  s2 += __shfl_xor(s2, 1); s2 += __shfl_xor(s2, 2);
  s2 += __shfl_xor(s2, 4); s2 += __shfl_xor(s2, 8);
  const float mu   = s * (1.0f / DIM);
  const float var  = s2 * (1.0f / DIM) - mu * mu;
  const float rstd = rsqrtf(var + 1e-5f);

  float* op = out + (size_t)(m0 + r) * DIM + seg * 32;
  const float* gp = gamma + seg * 32;
  const float* bp = beta + seg * 32;
  #pragma unroll
  for (int c = 0; c < 32; ++c) {
    float y = (rp[c] - mu) * rstd * gp[c] + bp[c];
    op[c] = elu1(y);
  }
}

// ---------------------------------------------------------------- launch
extern "C" void kernel_launch(void* const* d_in, const int* in_sizes, int n_in,
                              void* d_out, int out_size, void* d_ws, size_t ws_size,
                              hipStream_t stream) {
  const float*         x    = (const float*)d_in[0];
  const unsigned char* mask = (const unsigned char*)d_in[1];
  const float*         Wqkv = (const float*)d_in[2];
  const float*         bqkv = (const float*)d_in[3];
  const float*         Wout = (const float*)d_in[4];
  const float*         bout = (const float*)d_in[5];
  const float*         ln_g = (const float*)d_in[6];
  const float*         ln_b = (const float*)d_in[7];
  float* out = (float*)d_out;

  const size_t nx  = (size_t)M_ROWS * DIM;                       // 8388608
  const size_t nwq = (size_t)NQKV * DIM;                         // 786432
  const size_t nwo = (size_t)DIM * DIM;                          // 262144
  const size_t nhd = (size_t)BATCH * HEADS * SEQ * HEAD_DIM;     // 8388608

  char* ws = (char*)d_ws;
  u16* xbf  = (u16*)ws; ws += nx  * 2;
  u16* wqbf = (u16*)ws; ws += nwq * 2;
  u16* wobf = (u16*)ws; ws += nwo * 2;
  u16* qbf  = (u16*)ws; ws += nhd * 2;
  u16* kbf  = (u16*)ws; ws += nhd * 2;
  u16* vtbf = (u16*)ws; ws += nhd * 2;
  u16* aobf = (u16*)ws; ws += nx  * 2;                           // total ~82 MB

  cvt_f32_bf16<<<2048, 256, 0, stream>>>(x, xbf, (int)nx);
  cvt_f32_bf16<<<512, 256, 0, stream>>>(Wqkv, wqbf, (int)nwq);
  cvt_f32_bf16<<<256, 256, 0, stream>>>(Wout, wobf, (int)nwo);

  qkv_gemm<<<3072, 256, 0, stream>>>(xbf, wqbf, bqkv, qbf, kbf, vtbf);
  attention<<<1024, 256, 0, stream>>>(qbf, kbf, vtbf, mask, aobf);
  outproj_ln<<<1024, 256, 0, stream>>>(aobf, wobf, bout, ln_g, ln_b, out);
}
